// Transformer_61340722922022
// MI455X (gfx1250) — matrix operand order repaired and checked by reading, not compiled, not run on hardware
//
#include <hip/hip_runtime.h>
#include <hip/hip_bf16.h>
#include <cmath>

typedef __attribute__((ext_vector_type(16))) _Float16 v16h;
typedef __attribute__((ext_vector_type(8)))  _Float16 v8h;
typedef __attribute__((ext_vector_type(8)))  float    v8f;

#define EMBED    512
#define HEADS    8
#define HEAD_DIM 64
#define FF_DIM   2048
#define N_LAYERS 4
#define BATCH    4
#define SEQ      512           // SRC_LEN == TGT_LEN == 512
#define VOCAB    32000
#define MTOK     (BATCH * SEQ) // 2048 tokens

// ---------------------------------------------------------------------------
// Batched WMMA GEMM: C[m,n] = alpha * sum_k A[m,k]*B[n,k] (+bias) (+relu)
// A: f16 [M,K] row-major, B: f16 [N,K] row-major.
// Outputs: optional f32 C and/or f16 C16 (same geometry/strides).
// Batch index z = b*H + h with independent (b,h) strides.
// One wave computes a 16(M) x 64(N) strip: 4 v_wmma_f32_16x16x32_f16 / K-step,
// inner loop is pure global_load_b128 + wmma (no conversions).
// ---------------------------------------------------------------------------
struct GemmArgs {
  const _Float16* A; long long sAb, sAh; int lda;
  const _Float16* B; long long sBb, sBh; int ldb;
  float*          C;   long long sCb, sCh; int ldc;
  _Float16*       C16;
  const float* bias;
  int M, N, K, H;
  float alpha;
  int relu;
};

__global__ __launch_bounds__(128) void gemm_wmma_kernel(GemmArgs g) {
  const int z = blockIdx.z;
  const int b = z / g.H, h = z % g.H;
  const _Float16* A = g.A + (long long)b * g.sAb + (long long)h * g.sAh;
  const _Float16* B = g.B + (long long)b * g.sBb + (long long)h * g.sBh;
  const long long cOff = (long long)b * g.sCb + (long long)h * g.sCh;

  const int lane = threadIdx.x;
  const int lm   = lane & 15;
  const int hi   = lane >> 4;                    // lane half: 0 or 1
  const int m0   = (blockIdx.y * 4 + threadIdx.y) * 16;
  const int n0   = blockIdx.x * 64;
  if (m0 >= g.M) return;

  // A fragment: halves 0..7  = K[kk + hi*16 .. +7]
  //             halves 8..15 = K[kk + hi*16 + 8 .. +7]
  const _Float16* Arow = A + (long long)(m0 + lm) * g.lda + hi * 16;
  const _Float16* Brow[4];
#pragma unroll
  for (int t = 0; t < 4; ++t)
    Brow[t] = B + (long long)(n0 + t * 16 + lm) * g.ldb + hi * 16;

  v8f acc[4] = {v8f{}, v8f{}, v8f{}, v8f{}};

  for (int kk = 0; kk < g.K; kk += 32) {
    v8h a0 = *(const v8h*)(Arow + kk);
    v8h a1 = *(const v8h*)(Arow + kk + 8);
    v16h af;
#pragma unroll
    for (int i = 0; i < 8; ++i) {
      af[i]     = a0[i];
      af[8 + i] = a1[i];
    }
#pragma unroll
    for (int t = 0; t < 4; ++t) {
      // B frag: 16 contiguous halves, K[kk + hi*16 .. +15], column n per lane.
      v16h bf = *(const v16h*)(Brow[t] + kk);
      acc[t] = __builtin_amdgcn_wmma_f32_16x16x32_f16(
          false, af, false, bf, (short)0, acc[t], false, false);
    }
  }

#pragma unroll
  for (int t = 0; t < 4; ++t) {
    const int n = n0 + t * 16 + lm;
    const float bv = g.bias ? g.bias[n] : 0.0f;
#pragma unroll
    for (int r = 0; r < 8; ++r) {
      const int m = m0 + hi * 8 + r;           // C/D layout: vgpr r -> M=r(+8)
      float v = acc[t][r] * g.alpha + bv;
      if (g.relu) v = fmaxf(v, 0.0f);
      const long long idx = cOff + (long long)m * g.ldc + n;
      if (g.C)   g.C[idx]   = v;
      if (g.C16) g.C16[idx] = (_Float16)v;
    }
  }
}

// ---------------------------------------------------------------------------
// Weight transpose + convert: dst[c*R + r] = (f16) src[r*srcld + c]
// (f32 weights W[K,N] -> f16 [N,K])
// ---------------------------------------------------------------------------
__global__ __launch_bounds__(256) void wtrans_kernel(
    const float* src, int srcld, _Float16* dst, int R, int C) {
  long long i = (long long)blockIdx.x * 256 + threadIdx.x;
  if (i >= (long long)R * C) return;
  const int c = (int)(i % C);
  const long long r = i / C;
  dst[(long long)c * R + r] = (_Float16)src[r * srcld + c];
}

// f16 -> f16 batched slice transpose (V staging):
// dst[z][c*R + r] = src[b*sb + h*sh + r*srcld + c]
__global__ __launch_bounds__(256) void vtrans_kernel(
    const _Float16* src, long long sb, long long sh, int srcld,
    _Float16* dst, long long dstr, int R, int C, int H) {
  const int z = blockIdx.z;
  const int b = z / H, h = z % H;
  long long i = (long long)blockIdx.x * 256 + threadIdx.x;
  if (i >= (long long)R * C) return;
  const int c = (int)(i % C);
  const long long r = i / C;
  dst[(long long)z * dstr + (long long)c * R + r] =
      src[(long long)b * sb + (long long)h * sh + r * srcld + c];
}

// ---------------------------------------------------------------------------
// Mask + softmax over keys. Reference: energy = where(mask==0, -1e-10, energy)
// then softmax. keep = ids[b,k] != 0 && (!causal || k <= q).
// One wave per row; rows = B*H*Sq, Sk = 512. Writes f16 probabilities.
// ---------------------------------------------------------------------------
__global__ __launch_bounds__(256) void mask_softmax_kernel(
    const float* scores, _Float16* probs16, const int* ids,
    int Sq, int Sk, int H, int causal) {
  const int lane = threadIdx.x;
  const long long row = (long long)blockIdx.x * blockDim.y + threadIdx.y;
  const long long bh = row / Sq;
  const int q = (int)(row % Sq);
  const int b = (int)(bh / H);
  const float* p = scores + row * (long long)Sk;
  _Float16* po = probs16 + row * (long long)Sk;
  const int* id = ids + (long long)b * Sk;

  float v[16];
  float mx = -3.0e38f;
  const int NIT = Sk >> 5;                       // 16
  for (int i = 0; i < NIT; ++i) {
    const int k = i * 32 + lane;
    float x = p[k];
    const bool keep = (id[k] != 0) && (!causal || k <= q);
    x = keep ? x : -1e-10f;
    v[i] = x;
    mx = fmaxf(mx, x);
  }
#pragma unroll
  for (int o = 16; o > 0; o >>= 1) mx = fmaxf(mx, __shfl_xor(mx, o, 32));
  float sum = 0.0f;
  for (int i = 0; i < NIT; ++i) {
    const float e = expf(v[i] - mx);
    v[i] = e;
    sum += e;
  }
#pragma unroll
  for (int o = 16; o > 0; o >>= 1) sum += __shfl_xor(sum, o, 32);
  const float inv = 1.0f / sum;
  for (int i = 0; i < NIT; ++i) po[i * 32 + lane] = (_Float16)(v[i] * inv);
}

// ---------------------------------------------------------------------------
// LayerNorm: out = LN(x (+ res)) * g + b.  One wave per 512-wide row.
// Writes f32 (for residual chain) and optionally f16 (next GEMM A operand).
// ---------------------------------------------------------------------------
__global__ __launch_bounds__(256) void layernorm_kernel(
    const float* x, const float* res, const float* gw, const float* bw,
    float* out, _Float16* out16, int E) {
  const int lane = threadIdx.x;
  const long long row = (long long)blockIdx.x * blockDim.y + threadIdx.y;
  const float* xr = x + row * (long long)E;
  const float* rr = res ? res + row * (long long)E : nullptr;

  float v[16];
  float s = 0.0f, s2 = 0.0f;
  const int NIT = E >> 5;                        // 16
  for (int i = 0; i < NIT; ++i) {
    float t = xr[i * 32 + lane];
    if (rr) t += rr[i * 32 + lane];
    v[i] = t;
    s += t;
    s2 += t * t;
  }
#pragma unroll
  for (int o = 16; o > 0; o >>= 1) {
    s  += __shfl_xor(s,  o, 32);
    s2 += __shfl_xor(s2, o, 32);
  }
  const float mu  = s / (float)E;
  const float var = s2 / (float)E - mu * mu;
  const float rs  = rsqrtf(var + 1e-5f);
  for (int i = 0; i < NIT; ++i) {
    const int e = i * 32 + lane;
    const float o = (v[i] - mu) * rs * gw[e] + bw[e];
    out[row * (long long)E + e] = o;
    if (out16) out16[row * (long long)E + e] = (_Float16)o;
  }
}

// ---------------------------------------------------------------------------
// Embedding + sinusoidal PE.  pe_gated: only add PE where token != PAD (enc).
// ---------------------------------------------------------------------------
__global__ __launch_bounds__(256) void embed_kernel(
    const int* toks, const float* emb, float* out, _Float16* out16,
    int L, int E, float scale, int pe_gated) {
  const long long i = (long long)blockIdx.x * 256 + threadIdx.x;
  const int e = (int)(i % E);
  const long long t = i / E;
  const int s = (int)(t % L);
  const int tok = toks[t];
  float val = emb[(long long)tok * E + e] * scale;
  const float i2  = (float)(e & ~1);
  const float ang = (float)s / powf(10000.0f, i2 / (float)E);
  const float pe  = (e & 1) ? cosf(ang) : sinf(ang);
  if (!pe_gated || tok != 0) val += pe;
  out[i]   = val;
  out16[i] = (_Float16)val;
}

// ===========================================================================
// Host-side orchestration
// ===========================================================================
struct Lin { const float *b, *w; };
struct MHA { Lin k, o, q, v; };          // sorted-key order: k, o, q, v
struct LNP { const float *b, *g; };

struct WS {
  float *h, *d, *tmp, *scores;
  _Float16 *h16, *d16, *q16, *k16, *v16, *att16, *ff16, *probs16, *vt16, *wt16;
};

static inline void launch_gemm(hipStream_t st,
    const _Float16* A, long long sAb, long long sAh, int lda,
    const _Float16* B, long long sBb, long long sBh, int ldb,
    float* C, _Float16* C16, long long sCb, long long sCh, int ldc,
    const float* bias, int M, int N, int K, int nb, int nh,
    float alpha, int relu) {
  GemmArgs g{A, sAb, sAh, lda, B, sBb, sBh, ldb, C, sCb, sCh, ldc, C16,
             bias, M, N, K, nh, alpha, relu};
  dim3 grid((unsigned)(N / 64), (unsigned)(M / 64), (unsigned)(nb * nh));
  dim3 blk(32, 4, 1);
  gemm_wmma_kernel<<<grid, blk, 0, st>>>(g);
}

// trans weight W[K,N] (row stride N) -> f16 [N,K]
static inline void trans_weight(hipStream_t st, const float* w, int K, int N,
                                _Float16* dst) {
  const long long total = (long long)K * N;
  dim3 grid((unsigned)((total + 255) / 256), 1, 1);
  wtrans_kernel<<<grid, dim3(256, 1, 1), 0, st>>>(w, N, dst, K, N);
}

static inline void launch_ln(hipStream_t st, const float* x, const float* res,
                             const LNP& p, float* out, _Float16* out16) {
  dim3 grid(MTOK / 8, 1, 1), blk(32, 8, 1);
  layernorm_kernel<<<grid, blk, 0, st>>>(x, res, p.g, p.b, out, out16, EMBED);
}

// Full multi-head attention: out = O( softmax(mask(Q Kt / sqrt(E))) V )
static void run_attention(hipStream_t st, WS& w, const _Float16* q_in16,
                          const _Float16* kv_in16, const MHA& p,
                          const int* mask_ids, int causal, float* out) {
  const float scale = 1.0f / sqrtf((float)EMBED);

  // Q/K/V projections -> f16 activations
  trans_weight(st, p.q.w, EMBED, EMBED, w.wt16);
  launch_gemm(st, q_in16, 0, 0, EMBED, w.wt16, 0, 0, EMBED,
              nullptr, w.q16, 0, 0, EMBED, p.q.b,
              MTOK, EMBED, EMBED, 1, 1, 1.0f, 0);
  trans_weight(st, p.k.w, EMBED, EMBED, w.wt16);
  launch_gemm(st, kv_in16, 0, 0, EMBED, w.wt16, 0, 0, EMBED,
              nullptr, w.k16, 0, 0, EMBED, p.k.b,
              MTOK, EMBED, EMBED, 1, 1, 1.0f, 0);
  trans_weight(st, p.v.w, EMBED, EMBED, w.wt16);
  launch_gemm(st, kv_in16, 0, 0, EMBED, w.wt16, 0, 0, EMBED,
              nullptr, w.v16, 0, 0, EMBED, p.v.b,
              MTOK, EMBED, EMBED, 1, 1, 1.0f, 0);

  // scores[bh,q,k] = scale * Q . K   (K16 [token,512] sliced at h*64 is
  // already a valid [N,K]-layout f16 B operand -- no staging needed)
  launch_gemm(st,
      w.q16, (long long)SEQ * EMBED, HEAD_DIM, EMBED,
      w.k16, (long long)SEQ * EMBED, HEAD_DIM, EMBED,
      w.scores, nullptr, (long long)HEADS * SEQ * SEQ, (long long)SEQ * SEQ,
      SEQ, nullptr, SEQ, SEQ, HEAD_DIM, BATCH, HEADS, scale, 0);

  // mask + softmax -> f16 probabilities
  {
    dim3 grid((BATCH * HEADS * SEQ) / 8, 1, 1), blk(32, 8, 1);
    mask_softmax_kernel<<<grid, blk, 0, st>>>(w.scores, w.probs16, mask_ids,
                                              SEQ, SEQ, HEADS, causal);
  }

  // Stage V^T per (b,h) as f16 [64, token]
  {
    const long long total = (long long)SEQ * HEAD_DIM;
    dim3 grid((unsigned)((total + 255) / 256), 1, BATCH * HEADS);
    vtrans_kernel<<<grid, dim3(256, 1, 1), 0, st>>>(
        w.v16, (long long)SEQ * EMBED, HEAD_DIM, EMBED,
        w.vt16, (long long)HEAD_DIM * SEQ, SEQ, HEAD_DIM, HEADS);
  }

  // att[bh] @ V[bh] -> f16 [token, 512] with heads re-interleaved
  launch_gemm(st,
      w.probs16, (long long)HEADS * SEQ * SEQ, (long long)SEQ * SEQ, SEQ,
      w.vt16, (long long)HEADS * HEAD_DIM * SEQ, (long long)HEAD_DIM * SEQ, SEQ,
      nullptr, w.att16, (long long)SEQ * EMBED, HEAD_DIM, EMBED,
      nullptr, SEQ, HEAD_DIM, SEQ, BATCH, HEADS, 1.0f, 0);

  // output projection -> f32 (residual input)
  trans_weight(st, p.o.w, EMBED, EMBED, w.wt16);
  launch_gemm(st, w.att16, 0, 0, EMBED, w.wt16, 0, 0, EMBED,
              out, nullptr, 0, 0, EMBED, p.o.b,
              MTOK, EMBED, EMBED, 1, 1, 1.0f, 0);
}

static void run_ffn(hipStream_t st, WS& w, const _Float16* in16,
                    const Lin& ff1, const Lin& ff2, float* out) {
  trans_weight(st, ff1.w, EMBED, FF_DIM, w.wt16);
  launch_gemm(st, in16, 0, 0, EMBED, w.wt16, 0, 0, EMBED,
              nullptr, w.ff16, 0, 0, FF_DIM, ff1.b,
              MTOK, FF_DIM, EMBED, 1, 1, 1.0f, 1);
  trans_weight(st, ff2.w, FF_DIM, EMBED, w.wt16);
  launch_gemm(st, w.ff16, 0, 0, FF_DIM, w.wt16, 0, 0, FF_DIM,
              out, nullptr, 0, 0, EMBED, ff2.b,
              MTOK, EMBED, FF_DIM, 1, 1, 1.0f, 0);
}

extern "C" void kernel_launch(void* const* d_in, const int* in_sizes, int n_in,
                              void* d_out, int out_size, void* d_ws,
                              size_t ws_size, hipStream_t stream) {
  (void)in_sizes; (void)n_in; (void)out_size; (void)ws_size;

  // jax pytree (sorted-key) flattening of {'X','y','params'}:
  //   X, dec[0..3](26 each), emb_X, emb_y, enc[0..3](16 each), out.b, out.w, y
  auto F = [&](int i) { return (const float*)d_in[i]; };
  const int* X = (const int*)d_in[0];
  const float* embX  = F(105);
  const float* embY  = F(106);
  const float* out_b = F(171);
  const float* out_w = F(172);
  const int* Y = (const int*)d_in[173];

  // Workspace carve-up (256B aligned chunks)
  char* base = (char*)d_ws;
  size_t off = 0;
  auto take = [&](size_t bytes) -> void* {
    void* p = base + off;
    off += (bytes + 255) & ~(size_t)255;
    return p;
  };
  WS w;
  w.h       = (float*)take((size_t)MTOK * EMBED * 4);
  w.d       = (float*)take((size_t)MTOK * EMBED * 4);
  w.tmp     = (float*)take((size_t)MTOK * EMBED * 4);
  w.scores  = (float*)take((size_t)BATCH * HEADS * SEQ * SEQ * 4);
  w.h16     = (_Float16*)take((size_t)MTOK * EMBED * 2);
  w.d16     = (_Float16*)take((size_t)MTOK * EMBED * 2);
  w.q16     = (_Float16*)take((size_t)MTOK * EMBED * 2);
  w.k16     = (_Float16*)take((size_t)MTOK * EMBED * 2);
  w.v16     = (_Float16*)take((size_t)MTOK * EMBED * 2);
  w.att16   = (_Float16*)take((size_t)MTOK * EMBED * 2);
  w.ff16    = (_Float16*)take((size_t)MTOK * FF_DIM * 2);
  w.probs16 = (_Float16*)take((size_t)BATCH * HEADS * SEQ * SEQ * 2);
  w.vt16    = (_Float16*)take((size_t)BATCH * HEADS * SEQ * HEAD_DIM * 2);
  w.wt16    = (_Float16*)take((size_t)EMBED * VOCAB * 2);

  const float scale = sqrtf((float)EMBED);
  const unsigned embBlocks = (unsigned)(((size_t)MTOK * EMBED) / 256);

  // Embeddings + positional encodings (f32 + f16)
  embed_kernel<<<embBlocks, 256, 0, stream>>>(X, embX, w.h, w.h16,
                                              SEQ, EMBED, scale, 1);
  embed_kernel<<<embBlocks, 256, 0, stream>>>(Y, embY, w.d, w.d16,
                                              SEQ, EMBED, scale, 0);

  // ---------------- Encoder ----------------
  for (int l = 0; l < N_LAYERS; ++l) {
    const int b0 = 107 + 16 * l;
    MHA att = {{F(b0 + 0), F(b0 + 1)}, {F(b0 + 2), F(b0 + 3)},
               {F(b0 + 4), F(b0 + 5)}, {F(b0 + 6), F(b0 + 7)}};
    Lin ff1 = {F(b0 + 8), F(b0 + 9)};
    Lin ff2 = {F(b0 + 10), F(b0 + 11)};
    LNP ln1 = {F(b0 + 12), F(b0 + 13)};
    LNP ln2 = {F(b0 + 14), F(b0 + 15)};

    run_attention(stream, w, w.h16, w.h16, att, X, 0, w.tmp);
    launch_ln(stream, w.h, w.tmp, ln1, w.h, w.h16);
    run_ffn(stream, w, w.h16, ff1, ff2, w.tmp);
    launch_ln(stream, w.h, w.tmp, ln2, w.h, w.h16);
  }

  // ---------------- Decoder ----------------
  for (int l = 0; l < N_LAYERS; ++l) {
    const int b0 = 1 + 26 * l;
    MHA enc_att = {{F(b0 + 0), F(b0 + 1)}, {F(b0 + 2), F(b0 + 3)},
                   {F(b0 + 4), F(b0 + 5)}, {F(b0 + 6), F(b0 + 7)}};
    Lin ff1 = {F(b0 + 8), F(b0 + 9)};
    Lin ff2 = {F(b0 + 10), F(b0 + 11)};
    LNP ln1 = {F(b0 + 12), F(b0 + 13)};
    LNP ln2 = {F(b0 + 14), F(b0 + 15)};
    LNP ln3 = {F(b0 + 16), F(b0 + 17)};
    MHA self_att = {{F(b0 + 18), F(b0 + 19)}, {F(b0 + 20), F(b0 + 21)},
                    {F(b0 + 22), F(b0 + 23)}, {F(b0 + 24), F(b0 + 25)}};

    run_attention(stream, w, w.d16, w.d16, self_att, Y, 1, w.tmp);
    launch_ln(stream, w.d, w.tmp, ln1, w.d, w.d16);
    run_attention(stream, w, w.d16, w.h16, enc_att, X, 0, w.tmp);
    launch_ln(stream, w.d, w.tmp, ln2, w.d, w.d16);
    run_ffn(stream, w, w.d16, ff1, ff2, w.tmp);
    launch_ln(stream, w.tmp, nullptr, ln3, w.d, w.d16);  // ln3: NO residual
  }

  // ---------------- Output head ----------------
  trans_weight(stream, out_w, EMBED, VOCAB, w.wt16);
  launch_gemm(stream, w.d16, 0, 0, EMBED, w.wt16, 0, 0, EMBED,
              (float*)d_out, nullptr, 0, 0, VOCAB, out_b,
              MTOK, VOCAB, EMBED, 1, 1, 1.0f, 0);
}
